// VectorQuantizer_16750372454651
// MI455X (gfx1250) — matrix operand order repaired and checked by reading, not compiled, not run on hardware
//
#include <hip/hip_runtime.h>
#include <hip/hip_bf16.h>
#include <math.h>

// ---------------------------------------------------------------------------
// VQ-VAE vector quantizer for MI455X (gfx1250, wave32, WMMA).
//   z:         (16,256,32,32) fp32   -> zf (16384,256)
//   embedding: (16384,256)    fp32
// Core: fused GEMM(16384x16384x256) + row argmin using
// v_wmma_f32_16x16x32_bf16 with hi/lo bf16 split (3 products ~ fp32 argmin).
//  - 32 M-rows per wave (2 accumulation chains): 6 WMMAs per B-fragment load
//  - codebook double-buffered in LDS via GLOBAL_LOAD_ASYNC_TO_LDS_B128
//    (ASYNCcnt-tracked DMA, overlaps next-chunk staging with WMMA compute)
//  - codebook split in 2 halves across blocks (256 blocks) for WGP fill
// ---------------------------------------------------------------------------

typedef __bf16  v16bf __attribute__((ext_vector_type(16)));
typedef __bf16  v8bf  __attribute__((ext_vector_type(8)));
typedef float   v8f   __attribute__((ext_vector_type(8)));

#define N_VEC   16384      // flattened z vectors (16*32*32)
#define N_E     16384      // codebook entries
#define E_DIM   256
#define CHUNK   64         // codebook rows per LDS buffer (x2 buffers)
#define NHALF   (N_E / 2)  // codes per block (N split in 2)
#define BUF_ELT (CHUNK * E_DIM * 2)   // bf16 elements per buffer (hi+lo) = 32768

// ---------------- prep: zero counts -----------------------------------------
__global__ void k_zero(float* __restrict__ counts) {
    int i = blockIdx.x * 256 + threadIdx.x;
    if (i < N_E) counts[i] = 0.0f;
}

// ---------------- prep: transpose+convert z -> zf hi/lo bf16 -----------------
// grid (32 hw-tiles, 8 c-tiles, 16 b), block (32,32)
__global__ void __launch_bounds__(1024)
k_prep_z(const float* __restrict__ z, __bf16* __restrict__ zh, __bf16* __restrict__ zl) {
    __shared__ float tile[32][33];
    const int b = blockIdx.z, c0 = blockIdx.y * 32, hw0 = blockIdx.x * 32;
    const int tx = threadIdx.x, ty = threadIdx.y;
    tile[ty][tx] = z[(b * 256 + c0 + ty) * 1024 + hw0 + tx];   // coalesced (hw)
    __syncthreads();
    const int n = b * 1024 + hw0 + ty;
    const int c = c0 + tx;
    const float v  = tile[tx][ty];
    const __bf16 h = (__bf16)v;
    zh[n * E_DIM + c] = h;                                     // coalesced (c)
    zl[n * E_DIM + c] = (__bf16)(v - (float)h);
}

// ---------------- prep: convert embedding hi/lo bf16 + exact ||e||^2 ---------
__global__ void __launch_bounds__(256)
k_prep_e(const float* __restrict__ emb, __bf16* __restrict__ eh,
         __bf16* __restrict__ el, float* __restrict__ nrm) {
    __shared__ float red[256];
    const int j = blockIdx.x, c = threadIdx.x;
    const float v  = emb[j * E_DIM + c];
    const __bf16 h = (__bf16)v;
    eh[j * E_DIM + c] = h;
    el[j * E_DIM + c] = (__bf16)(v - (float)h);
    red[c] = v * v;
    __syncthreads();
    for (int s = 128; s > 0; s >>= 1) {
        if (c < s) red[c] += red[c + s];
        __syncthreads();
    }
    if (c == 0) nrm[j] = red[0];
}

// ---------------- async DMA: 32KB global -> LDS (no VGPR staging) ------------
// 128 threads, 16B/lane/op: 16 wave-level GLOBAL_LOAD_ASYNC_TO_LDS_B128 per
// 32KB region; INST-less addressing, addresses advance on both sides.
__device__ __forceinline__ void async_copy_32k(unsigned ldsDst, const void* gSrc, int tid) {
    unsigned lds = ldsDst + tid * 16;
    unsigned long long ga = (unsigned long long)(uintptr_t)gSrc + (unsigned long long)tid * 16;
    #pragma unroll
    for (int it = 0; it < 16; ++it) {
        asm volatile("global_load_async_to_lds_b128 %0, %1, off"
                     :: "v"(lds), "v"(ga) : "memory");
        lds += 2048;     // 128 threads * 16B
        ga  += 2048;
    }
}

// stage one codebook chunk (hi 32KB + lo 32KB) into an LDS buffer.
// ASYNCcnt += 32 per wave.
__device__ __forceinline__ void stage_chunk(unsigned ldsBuf,
                                            const __bf16* gH, const __bf16* gL, int tid) {
    async_copy_32k(ldsBuf,                 gH, tid);
    async_copy_32k(ldsBuf + CHUNK * E_DIM * 2 /*bytes: 32KB*/, gL, tid);
}

// ---------------- fused GEMM + argmin ---------------------------------------
// grid (128 M-blocks, 2 N-halves), 128 threads (4 waves). Each wave owns 32
// rows: A fragments (2 tiles x 8 K-steps, hi+lo) live in registers; codebook
// streamed through double-buffered 64KB LDS chunks via async DMA.
__global__ void __launch_bounds__(128)
k_argmin(const __bf16* __restrict__ zh, const __bf16* __restrict__ zl,
         const __bf16* __restrict__ eh, const __bf16* __restrict__ el,
         const float* __restrict__ nrmE,
         float* __restrict__ bvOut, int* __restrict__ biOut) {
    extern __shared__ __bf16 smem[];   // 2 buffers * 64KB

    const int tid  = threadIdx.x;
    const int wave = tid >> 5;
    const int lane = tid & 31;
    const int hi   = lane >> 4;           // 0: low half, 1: high half
    const int l16  = lane & 15;
    const int m0   = blockIdx.x * 128 + wave * 32;
    const int half = blockIdx.y;          // which 8192-code half
    const int jbase = half * NHALF;
    // low 32 bits of the flat shared pointer == LDS byte address
    const unsigned ldsBase = (unsigned)(uintptr_t)(void*)smem;

    // ---- preload A fragments for both 16-row tiles (ISA 16-bit A layout:
    //      low lanes hold K {t*32+0..7, t*32+16..23}, high lanes +8).
    v16bf aH0[8], aL0[8], aH1[8], aL1[8];
    {
        const int r0 = (m0 + l16) * E_DIM;
        const int r1 = (m0 + 16 + l16) * E_DIM;
        #pragma unroll
        for (int t = 0; t < 8; ++t) {
            const int k0 = t * 32 + hi * 8;
            v8bf p0 = *(const v8bf*)(zh + r0 + k0);
            v8bf p1 = *(const v8bf*)(zh + r0 + k0 + 16);
            v8bf q0 = *(const v8bf*)(zl + r0 + k0);
            v8bf q1 = *(const v8bf*)(zl + r0 + k0 + 16);
            v8bf s0 = *(const v8bf*)(zh + r1 + k0);
            v8bf s1 = *(const v8bf*)(zh + r1 + k0 + 16);
            v8bf u0 = *(const v8bf*)(zl + r1 + k0);
            v8bf u1 = *(const v8bf*)(zl + r1 + k0 + 16);
            #pragma unroll
            for (int i = 0; i < 8; ++i) {
                aH0[t][i] = p0[i]; aH0[t][8 + i] = p1[i];
                aL0[t][i] = q0[i]; aL0[t][8 + i] = q1[i];
                aH1[t][i] = s0[i]; aH1[t][8 + i] = s1[i];
                aL1[t][i] = u0[i]; aL1[t][8 + i] = u1[i];
            }
        }
    }

    float bestV[2][8];
    int   bestI[2][8];
    #pragma unroll
    for (int mt = 0; mt < 2; ++mt)
        #pragma unroll
        for (int r = 0; r < 8; ++r) { bestV[mt][r] = 3.4e38f; bestI[mt][r] = 0; }

    const int NCH = NHALF / CHUNK;   // 128 chunks of 64 codes
    // prologue: stage chunk 0 into buffer 0
    stage_chunk(ldsBase, eh + (size_t)jbase * E_DIM, el + (size_t)jbase * E_DIM, tid);

    for (int ci = 0; ci < NCH; ++ci) {
        // issue next chunk's DMA into the other buffer, then wait for the
        // current chunk (same-wave async loads complete in order).
        if (ci + 1 < NCH) {
            const size_t off = (size_t)(jbase + (ci + 1) * CHUNK) * E_DIM;
            stage_chunk(ldsBase + ((ci + 1) & 1) * (BUF_ELT * 2 /*bytes*/),
                        eh + off, el + off, tid);
            asm volatile("s_wait_asynccnt 32" ::: "memory");
        } else {
            asm volatile("s_wait_asynccnt 0" ::: "memory");
        }
        __syncthreads();   // current buffer visible to all waves

        const __bf16* sH = smem + (ci & 1) * BUF_ELT;
        const __bf16* sL = sH + CHUNK * E_DIM;
        const int jc = jbase + ci * CHUNK;

        #pragma unroll 1
        for (int nt = 0; nt < CHUNK / 16; ++nt) {
            const int n = nt * 16 + l16;   // B column owned by this lane
            v8f acc0 = {};
            v8f acc1 = {};
            #pragma unroll
            for (int t = 0; t < 8; ++t) {
                // B layout matching A: lane = column, K {0..7, 16..23}, high lanes +8
                const int boff = n * E_DIM + t * 32 + hi * 8;
                v8bf b0 = *(const v8bf*)(sH + boff);
                v8bf b1 = *(const v8bf*)(sH + boff + 16);
                v8bf c0 = *(const v8bf*)(sL + boff);
                v8bf c1 = *(const v8bf*)(sL + boff + 16);
                v16bf bH, bL;
                #pragma unroll
                for (int i = 0; i < 8; ++i) {
                    bH[i] = b0[i]; bH[8 + i] = b1[i];
                    bL[i] = c0[i]; bL[8 + i] = c1[i];
                }
                // two independent accumulation chains (ILP) sharing B frags
                acc0 = __builtin_amdgcn_wmma_f32_16x16x32_bf16(false, aH0[t], false, bH,
                                                               (short)0, acc0, false, false);
                acc1 = __builtin_amdgcn_wmma_f32_16x16x32_bf16(false, aH1[t], false, bH,
                                                               (short)0, acc1, false, false);
                acc0 = __builtin_amdgcn_wmma_f32_16x16x32_bf16(false, aH0[t], false, bL,
                                                               (short)0, acc0, false, false);
                acc1 = __builtin_amdgcn_wmma_f32_16x16x32_bf16(false, aH1[t], false, bL,
                                                               (short)0, acc1, false, false);
                acc0 = __builtin_amdgcn_wmma_f32_16x16x32_bf16(false, aL0[t], false, bH,
                                                               (short)0, acc0, false, false);
                acc1 = __builtin_amdgcn_wmma_f32_16x16x32_bf16(false, aL1[t], false, bH,
                                                               (short)0, acc1, false, false);
            }
            const int j  = jc + nt * 16 + l16;
            const float en = nrmE[j];
            #pragma unroll
            for (int r = 0; r < 8; ++r) {
                const float d0 = en - 2.0f * acc0[r];   // ||z||^2 constant per row
                const float d1 = en - 2.0f * acc1[r];
                if (d0 < bestV[0][r]) { bestV[0][r] = d0; bestI[0][r] = j; }
                if (d1 < bestV[1][r]) { bestV[1][r] = d1; bestI[1][r] = j; }
            }
        }
        __syncthreads();   // all waves done with this buffer before reuse
    }

    // min-reduce over the 16 lanes holding row m's N values
    #pragma unroll
    for (int mt = 0; mt < 2; ++mt) {
        #pragma unroll
        for (int r = 0; r < 8; ++r) {
            float v = bestV[mt][r];
            int   i = bestI[mt][r];
            #pragma unroll
            for (int off = 8; off >= 1; off >>= 1) {
                const float ov = __shfl_xor(v, off, 32);
                const int   oi = __shfl_xor(i, off, 32);
                if (ov < v || (ov == v && oi < i)) { v = ov; i = oi; }
            }
            if (l16 == 0) {
                const int m = m0 + mt * 16 + hi * 8 + r;   // D: M = r + 8*hi
                bvOut[half * N_VEC + m] = v;
                biOut[half * N_VEC + m] = i;
            }
        }
    }
}

// ---------------- merge the two N-halves ------------------------------------
__global__ void k_merge(const float* __restrict__ bv, const int* __restrict__ bi,
                        int* __restrict__ idx) {
    const int n = blockIdx.x * 256 + threadIdx.x;
    if (n < N_VEC) {
        const float v0 = bv[n], v1 = bv[N_VEC + n];
        // ties keep half 0 (lower j) -> matches argmin first-occurrence
        idx[n] = (v1 < v0) ? bi[N_VEC + n] : bi[n];
    }
}

// ---------------- gather + straight-through out + loss partials --------------
__global__ void __launch_bounds__(1024)
k_out(const float* __restrict__ z, const float* __restrict__ emb,
      const int* __restrict__ idx, float* __restrict__ out,
      float* __restrict__ partial) {
    __shared__ float tZ[32][33];
    __shared__ float tQ[32][33];
    __shared__ float red[1024];
    const int b = blockIdx.z, c0 = blockIdx.y * 32, hw0 = blockIdx.x * 32;
    const int tx = threadIdx.x, ty = threadIdx.y;
    const int tid = ty * 32 + tx;

    tZ[ty][tx] = z[(b * 256 + c0 + ty) * 1024 + hw0 + tx];   // coalesced
    __syncthreads();

    const int n = b * 1024 + hw0 + ty;
    const int j = idx[n];
    const float v = emb[j * E_DIM + c0 + tx];                // coalesced along c
    const float d = v - tZ[tx][ty];                          // zf[n][c0+tx]
    tQ[ty][tx] = v;
    red[tid]   = d * d;
    __syncthreads();

    out[(b * 256 + c0 + ty) * 1024 + hw0 + tx] = tQ[tx][ty]; // coalesced
    for (int s = 512; s > 0; s >>= 1) {
        if (tid < s) red[tid] += red[tid + s];
        __syncthreads();
    }
    if (tid == 0)
        partial[(blockIdx.z * gridDim.y + blockIdx.y) * gridDim.x + blockIdx.x] = red[0];
}

// ---------------- usage counts ----------------------------------------------
__global__ void k_counts(const int* __restrict__ idx, float* __restrict__ counts) {
    const int n = blockIdx.x * 256 + threadIdx.x;
    if (n < N_VEC) atomicAdd(&counts[idx[n]], 1.0f);   // integer-valued: exact
}

// ---------------- final scalars: loss + perplexity ---------------------------
__global__ void __launch_bounds__(256)
k_final(const float* __restrict__ partial, const float* __restrict__ counts,
        float* __restrict__ scalars) {
    __shared__ float red[256];
    float s = 0.0f;
    for (int i = threadIdx.x; i < 4096; i += 256) s += partial[i];
    red[threadIdx.x] = s;
    __syncthreads();
    for (int st = 128; st > 0; st >>= 1) {
        if (threadIdx.x < st) red[threadIdx.x] += red[threadIdx.x + st];
        __syncthreads();
    }
    const float loss = 1.25f * red[0] / 4194304.0f;   // (1+beta) * MSE
    __syncthreads();

    float e = 0.0f;
    for (int i = threadIdx.x; i < N_E; i += 256) {
        const float p = counts[i] * (1.0f / (float)N_VEC);
        e += p * logf(p + 1e-10f);
    }
    red[threadIdx.x] = e;
    __syncthreads();
    for (int st = 128; st > 0; st >>= 1) {
        if (threadIdx.x < st) red[threadIdx.x] += red[threadIdx.x + st];
        __syncthreads();
    }
    if (threadIdx.x == 0) {
        scalars[0] = loss;
        scalars[1] = expf(-red[0]);
    }
}

// ---------------- idx as float into output ----------------------------------
__global__ void k_idxf(const int* __restrict__ idx, float* __restrict__ o) {
    const int n = blockIdx.x * 256 + threadIdx.x;
    if (n < N_VEC) o[n] = (float)idx[n];
}

// ---------------------------------------------------------------------------
extern "C" void kernel_launch(void* const* d_in, const int* in_sizes, int n_in,
                              void* d_out, int out_size, void* d_ws, size_t ws_size,
                              hipStream_t stream) {
    const float* z   = (const float*)d_in[0];
    const float* emb = (const float*)d_in[1];
    float* out = (float*)d_out;

    // workspace layout (~33 MB)
    char* ws = (char*)d_ws;
    const size_t BF = (size_t)N_VEC * E_DIM * sizeof(__bf16);   // 8 MB
    __bf16* zh   = (__bf16*)(ws);
    __bf16* zl   = (__bf16*)(ws + BF);
    __bf16* eh   = (__bf16*)(ws + 2 * BF);
    __bf16* el   = (__bf16*)(ws + 3 * BF);
    char*   tail = ws + 4 * BF;
    float*  nrmE = (float*)(tail);                  // 64 KB
    int*    idx  = (int*)(tail + 65536);            // 64 KB
    float*  cnts = (float*)(tail + 2 * 65536);      // 64 KB
    float*  part = (float*)(tail + 3 * 65536);      // 16 KB
    float*  bv   = (float*)(tail + 3 * 65536 + 16384);          // 128 KB
    int*    bi   = (int*)(tail + 3 * 65536 + 16384 + 131072);   // 128 KB

    k_zero  <<<64, 256, 0, stream>>>(cnts);
    k_prep_z<<<dim3(32, 8, 16), dim3(32, 32), 0, stream>>>(z, zh, zl);
    k_prep_e<<<N_E, 256, 0, stream>>>(emb, eh, el, nrmE);
    k_argmin<<<dim3(128, 2), 128, 2 * BUF_ELT * sizeof(__bf16), stream>>>(
        zh, zl, eh, el, nrmE, bv, bi);
    k_merge <<<64, 256, 0, stream>>>(bv, bi, idx);
    k_out   <<<dim3(32, 8, 16), dim3(32, 32), 0, stream>>>(z, emb, idx, out, part);
    k_counts<<<64, 256, 0, stream>>>(idx, cnts);
    k_final <<<1, 256, 0, stream>>>(part, cnts, out + 4194304);
    k_idxf  <<<64, 256, 0, stream>>>(idx, out + 4194306);
}
